// MultiHeadAttention_36266703847923
// MI455X (gfx1250) — matrix operand order repaired and checked by reading, not compiled, not run on hardware
//
#include <hip/hip_runtime.h>
#include <hip/hip_bf16.h>

// ---------------------------------------------------------------------------
// MI455X / gfx1250 bf16-WMMA multi-head attention pipeline
// ---------------------------------------------------------------------------

typedef __attribute__((ext_vector_type(16))) __bf16       v16bf;
typedef __attribute__((ext_vector_type(8)))  float        v8f;
typedef __attribute__((ext_vector_type(4)))  unsigned int u32x4;
typedef __attribute__((ext_vector_type(4)))  int          v4i;

struct BFrag { u32x4 lo, hi; };   // 32 bytes == one 16-elem bf16 fragment slice

#define B_   4
#define SQ_  1024
#define SK_  1024
#define D_   1024
#define H_   16
#define DH_  64
#define HD_  (H_ * DH_)   // 1024

#if defined(__gfx1250__) && __has_builtin(__builtin_amdgcn_global_load_async_to_lds_b128) && __has_builtin(__builtin_amdgcn_s_wait_asynccnt)
#define USE_ASYNC_LDS 1
typedef __attribute__((address_space(1))) v4i* gv4_ptr;   // global int4*
typedef __attribute__((address_space(3))) v4i* lv4_ptr;   // LDS int4*
#else
#define USE_ASYNC_LDS 0
#endif

__device__ __forceinline__ __bf16 f2bf(float f) {
    unsigned int u = __builtin_bit_cast(unsigned int, f);
    unsigned int r = (u + 0x7FFFu + ((u >> 16) & 1u)) >> 16;   // RNE
    unsigned short s = (unsigned short)r;
    return __builtin_bit_cast(__bf16, s);
}

// pack two f32 (bit images) into one dword of two bf16 (RNE)
__device__ __forceinline__ unsigned int pk2bf(unsigned int lo, unsigned int hi) {
    unsigned int rl = ((lo + 0x7FFFu + ((lo >> 16) & 1u)) >> 16) & 0xFFFFu;
    unsigned int rh = (hi + 0x7FFFu + ((hi >> 16) & 1u)) & 0xFFFF0000u;
    return rl | rh;
}

__device__ __forceinline__ v8f wmma_bf16(v16bf a, v16bf b, v8f c) {
    return __builtin_amdgcn_wmma_f32_16x16x32_bf16(false, a, false, b,
                                                   (short)0, c, false, false);
}

__device__ __forceinline__ v16bf load_frag_g(const __bf16* p) {
    BFrag f;
    f.lo = *(const u32x4*)(p);
    f.hi = *(const u32x4*)(p + 8);
    return __builtin_bit_cast(v16bf, f);
}

// ---------------------------------------------------------------------------
// Stage 0a: f32 -> bf16, 8 elements per thread (b128 in, b128 out)
// ---------------------------------------------------------------------------
__global__ __launch_bounds__(256) void cvt_bf16_kernel(const float* __restrict__ in,
                                                       __bf16* __restrict__ out, int n) {
    int i = (blockIdx.x * 256 + threadIdx.x) * 8;
    if (i >= n) return;
    u32x4 a = *(const u32x4*)(in + i);
    u32x4 b = *(const u32x4*)(in + i + 4);
    u32x4 r;
    r.x = pk2bf(a.x, a.y);
    r.y = pk2bf(a.z, a.w);
    r.z = pk2bf(b.x, b.y);
    r.w = pk2bf(b.z, b.w);
    *(u32x4*)(out + i) = r;
}

// ---------------------------------------------------------------------------
// Stage 0b: weights f32 [K,N] -> WMMA-B-fragment-ready bf16 layout
//   frag(kb=k/32, nt=n/16): lane = (n%16) + 16*((k%32)/16), slot = k%16
// ---------------------------------------------------------------------------
__global__ __launch_bounds__(256) void cvt_wfrag_kernel(const float* __restrict__ in,
                                                        __bf16* __restrict__ out,
                                                        int K, int N) {
    int i = blockIdx.x * 256 + threadIdx.x;
    if (i >= K * N) return;
    int k = i / N, n = i - k * N;
    int kb = k >> 5, kc = k & 31, nt = n >> 4, nn = n & 15;
    int lane = nn + ((kc >> 4) << 4);
    int slot = kc & 15;
    size_t addr = (((size_t)kb * (N >> 4) + nt) << 9) + (size_t)(lane << 4) + slot;
    out[addr] = f2bf(in[i]);
}

// ---------------------------------------------------------------------------
// Stage 1: GEMM  C[M,N] = A[M,K](bf16,row-major) x Bfrag + bias
// 256 threads (8 waves, 4x2), tile 128x128, BK=32.
// A tile double-buffered in LDS via GLOBAL_LOAD_ASYNC_TO_LDS_B128 (ASYNCcnt).
// MODE: 0 bf16 row-major, 1 f32 row-major, 2 K^T attn-frag, 3 V attn-frag
// ---------------------------------------------------------------------------
template <int MODE>
__device__ __forceinline__ void store_elem(void* out, int row, int col, int N, float v) {
    if constexpr (MODE == 0) {
        ((__bf16*)out)[(size_t)row * N + col] = f2bf(v);
    } else if constexpr (MODE == 1) {
        ((float*)out)[(size_t)row * N + col] = v;
    } else if constexpr (MODE == 2) {
        // K^T fragments: contraction = d (head dim), n = key position
        int b = row >> 10, kpos = row & 1023;
        int h = col >> 6,  d    = col & 63;
        int blk = b * H_ + h;
        int lane = (kpos & 15) + (((d & 31) >> 4) << 4);
        int slot = d & 15;
        size_t addr = ((((size_t)blk * (SK_ >> 4) + (kpos >> 4)) * 2 + (d >> 5)) << 9)
                      + (size_t)(lane << 4) + slot;
        ((__bf16*)out)[addr] = f2bf(v);
    } else {
        // V fragments: contraction = key position, n = d (head dim)
        int b = row >> 10, kpos = row & 1023;
        int h = col >> 6,  d    = col & 63;
        int blk = b * H_ + h;
        int lane = (d & 15) + (((kpos & 31) >> 4) << 4);
        int slot = kpos & 15;
        size_t addr = ((((size_t)blk * (SK_ >> 5) + (kpos >> 5)) * 4 + (d >> 4)) << 9)
                      + (size_t)(lane << 4) + slot;
        ((__bf16*)out)[addr] = f2bf(v);
    }
}

template <int MODE>
__global__ __launch_bounds__(256) void gemm_kernel(const __bf16* __restrict__ A,
                                                   const __bf16* __restrict__ Bf,
                                                   const float*  __restrict__ bias,
                                                   void* __restrict__ Out,
                                                   int M, int N, int K) {
    __shared__ __bf16 As[2][128 * 40];   // double buffer; padded stride 40

    const int tid  = threadIdx.x;
    const int lane = tid & 31;
    const int wave = tid >> 5;
    const int wm   = wave & 3;
    const int wn   = wave >> 2;
    const int bm0  = blockIdx.y * 128;
    const int bn0  = blockIdx.x * 128;

    const int lm = lane & 15, lh = lane >> 4;
    const int arow0 = wm * 32 + lm;

    // cooperative A-tile loader coordinates (each thread: 32B of one row)
    const int ldr = tid >> 1;
    const int ldc = (tid & 1) << 4;
    const __bf16* Abase = A + (size_t)(bm0 + ldr) * K + ldc;

    v8f acc[2][4];
#pragma unroll
    for (int i = 0; i < 2; ++i)
#pragma unroll
        for (int j = 0; j < 4; ++j) acc[i][j] = v8f{};

    const int nsteps = K >> 5;

    // stage tile 0
#if USE_ASYNC_LDS
    {
        __bf16* lp = &As[0][ldr * 40 + ldc];
        __builtin_amdgcn_global_load_async_to_lds_b128((gv4_ptr)Abase, (lv4_ptr)lp, 0, 0);
        __builtin_amdgcn_global_load_async_to_lds_b128((gv4_ptr)(Abase + 8),
                                                       (lv4_ptr)(lp + 8), 0, 0);
        __builtin_amdgcn_s_wait_asynccnt(0);
    }
#else
    {
        u32x4 d0 = *(const u32x4*)(Abase);
        u32x4 d1 = *(const u32x4*)(Abase + 8);
        __bf16* lp = &As[0][ldr * 40 + ldc];
        *(u32x4*)(lp)     = d0;
        *(u32x4*)(lp + 8) = d1;
    }
#endif
    __syncthreads();

    for (int kb = 0; kb < nsteps; ++kb) {
        const int cur = kb & 1;
        const bool have_next = (kb + 1) < nsteps;

#if USE_ASYNC_LDS
        if (have_next) {   // async copy of next tile overlaps this tile's WMMAs
            const __bf16* gp = Abase + (kb + 1) * 32;
            __bf16* lp = &As[cur ^ 1][ldr * 40 + ldc];
            __builtin_amdgcn_global_load_async_to_lds_b128((gv4_ptr)gp, (lv4_ptr)lp, 0, 0);
            __builtin_amdgcn_global_load_async_to_lds_b128((gv4_ptr)(gp + 8),
                                                           (lv4_ptr)(lp + 8), 0, 0);
        }
#else
        u32x4 n0, n1;
        if (have_next) {
            const __bf16* gp = Abase + (kb + 1) * 32;
            n0 = *(const u32x4*)(gp);
            n1 = *(const u32x4*)(gp + 8);
        }
#endif

        // A fragments from LDS
        v16bf afr[2];
#pragma unroll
        for (int mi = 0; mi < 2; ++mi) {
            const __bf16* lp = &As[cur][(arow0 + mi * 16) * 40];
            BFrag f;
            f.lo = *(const u32x4*)(lp + 16 * lh);
            f.hi = *(const u32x4*)(lp + 16 * lh + 8);
            afr[mi] = __builtin_bit_cast(v16bf, f);
        }

        // all four B fragments into distinct regs first (one clause, staggered waits)
        v16bf bfr[4];
#pragma unroll
        for (int nj = 0; nj < 4; ++nj) {
            int nt = (bn0 >> 4) + wn * 4 + nj;
            bfr[nj] = load_frag_g(Bf + (((size_t)kb * (N >> 4) + nt) << 9) + (lane << 4));
        }
#pragma unroll
        for (int nj = 0; nj < 4; ++nj) {
            acc[0][nj] = wmma_bf16(afr[0], bfr[nj], acc[0][nj]);
            acc[1][nj] = wmma_bf16(afr[1], bfr[nj], acc[1][nj]);
        }

#if USE_ASYNC_LDS
        if (have_next) __builtin_amdgcn_s_wait_asynccnt(0);
#else
        if (have_next) {
            __bf16* lp = &As[cur ^ 1][ldr * 40 + ldc];
            *(u32x4*)(lp)     = n0;
            *(u32x4*)(lp + 8) = n1;
        }
#endif
        __syncthreads();
    }

    // epilogue: bias + mode-specific store
#pragma unroll
    for (int nj = 0; nj < 4; ++nj) {
        int col  = bn0 + wn * 64 + nj * 16 + lm;
        float bv = bias[col];
#pragma unroll
        for (int mi = 0; mi < 2; ++mi) {
#pragma unroll
            for (int r = 0; r < 8; ++r) {
                int row = bm0 + wm * 32 + mi * 16 + r + 8 * lh;
                store_elem<MODE>(Out, row, col, N, acc[mi][nj][r] + bv);
            }
        }
    }
}

// ---------------------------------------------------------------------------
// Stage 2: flash attention, one wave per (b, h, 16-query tile).
// Tile-wide running max (exact softmax for any shared M); denominator via
// ones-column WMMA. K frags batched before S-WMMAs; V frags issued before
// the softmax VALU so their latency hides under exp/max work.
// ---------------------------------------------------------------------------
__global__ __launch_bounds__(32) void attn_kernel(const __bf16* __restrict__ Qb,
                                                  const __bf16* __restrict__ Kf,
                                                  const __bf16* __restrict__ Vf,
                                                  const unsigned char* __restrict__ vmask,
                                                  const unsigned char* __restrict__ qmask,
                                                  __bf16* __restrict__ Ctx) {
    __shared__ __bf16 Ps[512];   // P tile staged in A-fragment layout

    const int lane = threadIdx.x;
    const int lm = lane & 15, lh = lane >> 4;
    const int qt = blockIdx.x;
    const int h  = blockIdx.y;
    const int b  = blockIdx.z;
    const int q0 = qt * 16;
    const int blk = b * H_ + h;

    // Q fragments (A-layout) direct from global
    v16bf qf[2];
    {
        const __bf16* qp = Qb + (size_t)(b * SQ_ + q0 + lm) * HD_ + h * DH_;
#pragma unroll
        for (int ks = 0; ks < 2; ++ks) {
            BFrag f;
            f.lo = *(const u32x4*)(qp + ks * 32 + 16 * lh);
            f.hi = *(const u32x4*)(qp + ks * 32 + 16 * lh + 8);
            qf[ks] = __builtin_bit_cast(v16bf, f);
        }
    }

    // ones B-fragment: column 0 of the result accumulates row sums
    v16bf onesf;
    {
        unsigned int ob = (lm == 0) ? 0x3F803F80u : 0u;   // bf16 1.0 pair
        BFrag f;
        f.lo = u32x4{ob, ob, ob, ob};
        f.hi = u32x4{ob, ob, ob, ob};
        onesf = __builtin_bit_cast(v16bf, f);
    }

    v8f o[4], ol;
#pragma unroll
    for (int i = 0; i < 4; ++i) o[i] = v8f{};
    ol = v8f{};
    float rm = -3.0e38f;   // tile-wide running max (wave-uniform)

    const int niter = SK_ >> 5;
    for (int it = 0; it < niter; ++it) {
        // batch all four K^T fragments (distinct regs -> one clause)
        v16bf kfr[4];
#pragma unroll
        for (int j = 0; j < 2; ++j)
#pragma unroll
            for (int ks = 0; ks < 2; ++ks)
                kfr[j * 2 + ks] = load_frag_g(
                    Kf + ((((size_t)blk * (SK_ >> 4) + (it * 2 + j)) * 2 + ks) << 9)
                       + (lane << 4));

        v8f s[2];
        s[0] = v8f{}; s[1] = v8f{};
#pragma unroll
        for (int j = 0; j < 2; ++j)
#pragma unroll
            for (int ks = 0; ks < 2; ++ks)
                s[j] = wmma_bf16(qf[ks], kfr[j * 2 + ks], s[j]);

        // issue V fragments now: latency overlaps the softmax VALU below
        v16bf vfr[4];
#pragma unroll
        for (int dt = 0; dt < 4; ++dt)
            vfr[dt] = load_frag_g(
                Vf + ((((size_t)blk * (SK_ >> 5) + it) * 4 + dt) << 9) + (lane << 4));

        // key mask (unscaled scores, -1e9 fill per Keras Attention)
        float mk0 = vmask[b * SK_ + it * 32 + lm]      ? 0.f : -1.0e9f;
        float mk1 = vmask[b * SK_ + it * 32 + 16 + lm] ? 0.f : -1.0e9f;

        float e0[8], e1[8];
        float mx = -3.0e38f;
#pragma unroll
        for (int r = 0; r < 8; ++r) {
            e0[r] = s[0][r] + mk0;
            e1[r] = s[1][r] + mk1;
            mx = fmaxf(mx, fmaxf(e0[r], e1[r]));
        }
#pragma unroll
        for (int m = 1; m < 32; m <<= 1) mx = fmaxf(mx, __shfl_xor(mx, m, 32));

        if (mx > rm) {   // uniform branch; rarely taken once max stabilizes
            float corr = __expf(rm - mx);
            rm = mx;
#pragma unroll
            for (int dt = 0; dt < 4; ++dt) o[dt] *= corr;
            ol *= corr;
        }

        __syncthreads();   // single-wave WG: S_NOP, orders LDS reuse
        // P = exp(S - rm), transposed (C-layout -> A-fragment layout) via LDS
#pragma unroll
        for (int j = 0; j < 2; ++j) {
#pragma unroll
            for (int r = 0; r < 8; ++r) {
                int m  = r + 8 * lh;
                int k  = j * 16 + lm;
                int hh = k >> 4;
                int e  = k & 15;
                float pv = __expf((j == 0 ? e0[r] : e1[r]) - rm);
                Ps[(m + 16 * hh) * 16 + e] = f2bf(pv);
            }
        }
        __syncthreads();

        BFrag pfr;
        pfr.lo = *(const u32x4*)&Ps[lane << 4];
        pfr.hi = *(const u32x4*)&Ps[(lane << 4) + 8];
        v16bf pf = __builtin_bit_cast(v16bf, pfr);

        // O += P x V ; denominator via ones-column WMMA
#pragma unroll
        for (int dt = 0; dt < 4; ++dt) o[dt] = wmma_bf16(pf, vfr[dt], o[dt]);
        ol = wmma_bf16(pf, onesf, ol);
    }

    // epilogue: normalize by broadcast denominator, q_mask zeroing, store ctx
#pragma unroll
    for (int r = 0; r < 8; ++r) {
        int m   = r + 8 * lh;
        int row = b * SQ_ + q0 + m;
        float denom = __shfl(ol[r], 16 * lh, 32);   // col-0 lane of this half
        float scale = (denom > 0.f ? 1.f / denom : 0.f)
                    * (qmask[b * SQ_ + q0 + m] ? 1.f : 0.f);
#pragma unroll
        for (int dt = 0; dt < 4; ++dt) {
            Ctx[(size_t)row * HD_ + h * DH_ + dt * 16 + lm] = f2bf(o[dt][r] * scale);
        }
    }
}

// ---------------------------------------------------------------------------
// Host launcher
// ---------------------------------------------------------------------------
extern "C" void kernel_launch(void* const* d_in, const int* in_sizes, int n_in,
                              void* d_out, int out_size, void* d_ws, size_t ws_size,
                              hipStream_t stream) {
    const float* q  = (const float*)d_in[0];
    const float* v  = (const float*)d_in[1];
    const unsigned char* qmask = (const unsigned char*)d_in[2];
    const unsigned char* vmask = (const unsigned char*)d_in[3];
    const float* Wq = (const float*)d_in[4];  const float* bq = (const float*)d_in[5];
    const float* Wk = (const float*)d_in[6];  const float* bk = (const float*)d_in[7];
    const float* Wv = (const float*)d_in[8];  const float* bv = (const float*)d_in[9];
    const float* Wo = (const float*)d_in[10]; const float* bo = (const float*)d_in[11];

    char* p = (char*)d_ws;
    auto carve = [&](size_t bytes) { char* r = p; p += bytes; return r; };
    const size_t ACT = (size_t)B_ * SQ_ * D_ * sizeof(__bf16);   // 8 MB
    const size_t WGT = (size_t)D_ * HD_ * sizeof(__bf16);        // 2 MB

    __bf16* qb  = (__bf16*)carve(ACT);
    __bf16* vb  = (__bf16*)carve(ACT);
    __bf16* wqf = (__bf16*)carve(WGT);
    __bf16* wkf = (__bf16*)carve(WGT);
    __bf16* wvf = (__bf16*)carve(WGT);
    __bf16* wof = (__bf16*)carve(WGT);
    __bf16* Qb  = (__bf16*)carve(ACT);
    __bf16* Kfr = (__bf16*)carve(ACT);
    __bf16* Vfr = (__bf16*)carve(ACT);
    __bf16* ctx = vb;   // vb is dead after the K/V projection GEMMs

    const int nAct = B_ * SQ_ * D_;
    const int nWgt = D_ * HD_;

    cvt_bf16_kernel<<<(nAct / 8 + 255) / 256, 256, 0, stream>>>(q, qb, nAct);
    cvt_bf16_kernel<<<(nAct / 8 + 255) / 256, 256, 0, stream>>>(v, vb, nAct);
    cvt_wfrag_kernel<<<(nWgt + 255) / 256, 256, 0, stream>>>(Wq, wqf, D_, HD_);
    cvt_wfrag_kernel<<<(nWgt + 255) / 256, 256, 0, stream>>>(Wk, wkf, D_, HD_);
    cvt_wfrag_kernel<<<(nWgt + 255) / 256, 256, 0, stream>>>(Wv, wvf, D_, HD_);
    cvt_wfrag_kernel<<<(nWgt + 255) / 256, 256, 0, stream>>>(Wo, wof, HD_, D_);

    dim3 ggrid(HD_ / 128, (B_ * SQ_) / 128);   // (8, 32)
    gemm_kernel<0><<<ggrid, 256, 0, stream>>>(qb, wqf, bq, (void*)Qb,  B_ * SQ_, HD_, D_);
    gemm_kernel<2><<<ggrid, 256, 0, stream>>>(vb, wkf, bk, (void*)Kfr, B_ * SK_, HD_, D_);
    gemm_kernel<3><<<ggrid, 256, 0, stream>>>(vb, wvf, bv, (void*)Vfr, B_ * SK_, HD_, D_);

    dim3 agrid(SQ_ / 16, H_, B_);
    attn_kernel<<<agrid, 32, 0, stream>>>(Qb, Kfr, Vfr, vmask, qmask, ctx);

    gemm_kernel<1><<<ggrid, 256, 0, stream>>>(ctx, wof, bo, d_out, B_ * SQ_, D_, HD_);
}
